// RNN_72464688218603
// MI455X (gfx1250) — matrix operand order repaired and checked by reading, not compiled, not run on hardware
//
#include <hip/hip_runtime.h>
#include <hip/hip_bf16.h>
#include <math.h>

// Problem sizes (match reference)
#define BZ   64
#define TT   2048
#define IDIM 1024
#define HDIM 1024
#define ODIM 512

typedef __attribute__((ext_vector_type(16))) __bf16 v16bf;
typedef __attribute__((ext_vector_type(16))) float  v16f;
typedef __attribute__((ext_vector_type(8)))  float  v8f;

// ---------- scalar converts (hardware v_cvt_pk_bf16_f32, RNE) ----------
__device__ __forceinline__ unsigned short f2bf(float f) {
    union { __bf16 h; unsigned short u; } c; c.h = (__bf16)f; return c.u;
}
__device__ __forceinline__ float bf2f(unsigned short s) {
    union { unsigned short u; __bf16 h; } c; c.u = s; return (float)c.h;
}

// ---------- WMMA fragment loaders (wave32; contiguous K = 16*(lane/16) + j in both A and B) ----------
// B fragment (32x16 KxN): lane = column n = lane%16; 16 contiguous K bf16 at p.
__device__ __forceinline__ v16bf load_bfrag(const unsigned short* p) {
    union { uint4 q[2]; v16bf v; } u;
    u.q[0] = *(const uint4*)(p);
    u.q[1] = *(const uint4*)(p + 8);
    return u.v;
}
// A fragment (16x32 bf16): lane row m = lane%16; halves 0..15 <-> K=k0+16g+j
// (g = lane/16).
__device__ __forceinline__ v16bf load_afrag_bf16(const unsigned short* rowbase,
                                                 int k0, int g) {
    union { uint4 q[2]; v16bf v; } u;
    u.q[0] = *(const uint4*)(rowbase + k0 + 16 * g);
    u.q[1] = *(const uint4*)(rowbase + k0 + 16 * g + 8);
    return u.v;
}
// A fragment sourced from fp32 memory -> hardware packed converts.
__device__ __forceinline__ v16bf load_afrag_f32(const float* rowbase, int k0, int g) {
    union { float4 q[4]; v16f v; } u;
    u.q[0] = *(const float4*)(rowbase + k0 + 16 * g);
    u.q[1] = *(const float4*)(rowbase + k0 + 16 * g + 4);
    u.q[2] = *(const float4*)(rowbase + k0 + 16 * g + 8);
    u.q[3] = *(const float4*)(rowbase + k0 + 16 * g + 12);
    return __builtin_convertvector(u.v, v16bf);
}

// ---------- CDNA5 async copy: global -> LDS (ASYNCcnt tracked) ----------
__device__ __forceinline__ void async_load_b128(unsigned lds_off, const void* gptr) {
    asm volatile("global_load_async_to_lds_b128 %0, %1, off"
                 :: "v"(lds_off), "v"((unsigned long long)(uintptr_t)gptr)
                 : "memory");
}
__device__ __forceinline__ void wait_async0() {
    asm volatile("s_wait_asynccnt 0x0" ::: "memory");
}

// ---------- small utility kernels ----------
__global__ void cvt_f32_to_bf16(const float* __restrict__ src,
                                unsigned short* __restrict__ dst, int n) {
    int i = blockIdx.x * 256 + threadIdx.x;
    if (i < n) dst[i] = f2bf(src[i]);
}
__global__ void zero_u16(unsigned short* __restrict__ dst, int n) {
    int i = blockIdx.x * 256 + threadIdx.x;
    if (i < n) dst[i] = 0;
}
__global__ void zero_u32(unsigned int* __restrict__ dst, int n) {
    int i = blockIdx.x * 256 + threadIdx.x;
    if (i < n) dst[i] = 0;
}

// ---------- Phase 1: XW[t*64+b][h] = x[b,t,:] @ W_ih[h,:] + b_h[h]  (bf16 out) ----
// Grid (16 nblocks x 1024 mblocks), 256 thr (8 waves). Wave w: M-tile w x 4 N-tiles.
// W_ih tiles shared by all 8 waves -> double-buffered async-LDS staging.
#define P1PAD 40   // LDS row stride in ushorts (32 data + 8 pad -> disjoint banks)
__global__ void __launch_bounds__(256)
xw_gemm(const float* __restrict__ X, const unsigned short* __restrict__ Wih,
        const float* __restrict__ bh, unsigned short* __restrict__ XW) {
    __shared__ unsigned short bstage[2][64 * P1PAD];
    const int tid  = threadIdx.x;
    const int lane = tid & 31, wave = tid >> 5;
    const int g = lane >> 4, ln = lane & 15;
    const int n0    = blockIdx.x * 64;
    const int Rbase = blockIdx.y * 128 + wave * 16;

    // A row for this lane: global GEMM row R = t*64 + b  ->  x[b][t][:]
    const int R = Rbase + ln;
    const int t = R >> 6, b = R & 63;
    const float* xrow = X + ((size_t)b * TT + t) * IDIM;

    // async staging: thread covers (row = tid/4) x (16B segment = tid%4) of a
    // 64-row x 32-K bf16 chunk of W_ih (rows n0..n0+63)
    const int srow = tid >> 2, sseg = tid & 3;
    const unsigned short* gsrc = Wih + (size_t)(n0 + srow) * IDIM + sseg * 8;
    const unsigned ldsdst[2] = {
        (unsigned)(uintptr_t)&bstage[0][srow * P1PAD + sseg * 8],
        (unsigned)(uintptr_t)&bstage[1][srow * P1PAD + sseg * 8] };

    async_load_b128(ldsdst[0], gsrc);
    wait_async0();
    __syncthreads();

    v8f acc[4] = {};
    int cur = 0;
    for (int k0 = 0; k0 < IDIM; k0 += 32) {
        if (k0 + 32 < IDIM) {
            async_load_b128(ldsdst[cur ^ 1], gsrc + k0 + 32);
            __builtin_prefetch(xrow + k0 + 32, 0, 3);  // next A chunk toward L2/WGP$
        }
        // batch all fragment loads first (one clause, one dscnt wait) ...
        v16bf a = load_afrag_f32(xrow, k0, g);
        v16bf bm[4];
#pragma unroll
        for (int nt = 0; nt < 4; ++nt)
            bm[nt] = load_bfrag(&bstage[cur][(nt * 16 + ln) * P1PAD + 16 * g]);
        // ... then issue the WMMAs back-to-back
#pragma unroll
        for (int nt = 0; nt < 4; ++nt)
            acc[nt] = __builtin_amdgcn_wmma_f32_16x16x32_bf16(
                false, a, false, bm[nt], (short)0, acc[nt], false, false);
        wait_async0();
        __syncthreads();
        cur ^= 1;
    }
#pragma unroll
    for (int nt = 0; nt < 4; ++nt) {
        const int col    = n0 + nt * 16 + ln;
        const float bias = bh[col];
#pragma unroll
        for (int r = 0; r < 8; ++r) {
            const int row = Rbase + r + 8 * g;
            XW[(size_t)row * HDIM + col] = f2bf(acc[nt][r] + bias);
        }
    }
}

// ---------- Phase 2: persistent scan kernel -----------------------------------
// ONE kernel runs all 2048 steps. 16 workgroups (one per 64-col slice), each
// stages its W_hh slice (64x1024 bf16, padded, ~129KB of the WGP's 320KB LDS)
// ONCE, then loops: Hout = tanh(XW_t + Hin @ Whh_slice^T); device barrier; swap.
// Only 16 WGs exist -> per-wave ILP is the latency hider: K loop unrolled x8.
#define HPAD 1032  // LDS row stride in ushorts (1024 data + 8 pad)
__global__ void __launch_bounds__(256)
rnn_scan(const unsigned short* __restrict__ Whh,
         const unsigned short* __restrict__ XW,
         unsigned short* __restrict__ Hbuf,
         unsigned int* __restrict__ cnt) {
    __shared__ unsigned short wstage[64 * HPAD];
    const int tid  = threadIdx.x;
    const int lane = tid & 31, wave = tid >> 5;
    const int g = lane >> 4, ln = lane & 15;
    const int mt = wave & 3, ntb = (wave >> 2) * 2;
    const int n0 = blockIdx.x * 64;

    // stage W_hh rows [n0, n0+64) x all K into LDS (once for the whole scan)
    for (int idx = tid; idx < 64 * 128; idx += 256) {
        const int row = idx >> 7, seg = idx & 127;
        *(uint4*)&wstage[row * HPAD + seg * 8] =
            *(const uint4*)&Whh[(size_t)(n0 + row) * HDIM + seg * 8];
    }
    __syncthreads();

    const unsigned short* bp0 = &wstage[((ntb + 0) * 16 + ln) * HPAD + 16 * g];
    const unsigned short* bp1 = &wstage[((ntb + 1) * 16 + ln) * HPAD + 16 * g];

    for (int t = 0; t < TT; ++t) {
        const unsigned short* Hin  = Hbuf + (size_t)(t & 1) * (BZ * HDIM);
        unsigned short*       Hout = Hbuf + (size_t)((t + 1) & 1) * (BZ * HDIM);
        const unsigned short* XWt  = XW + (size_t)t * (BZ * HDIM);
        const unsigned short* arow = Hin + (size_t)(mt * 16 + ln) * HDIM;

        // pull this step's XW addend toward the caches while the K loop runs
        __builtin_prefetch(XWt + (size_t)(mt * 16 + 8 * g) * HDIM + n0 + ntb * 16 + ln, 0, 3);

        v8f acc[2] = {};
#pragma unroll 8
        for (int k0 = 0; k0 < HDIM; k0 += 32) {
            v16bf a  = load_afrag_bf16(arow, k0, g);
            v16bf b0 = load_bfrag(bp0 + k0);
            v16bf b1 = load_bfrag(bp1 + k0);
            acc[0] = __builtin_amdgcn_wmma_f32_16x16x32_bf16(
                false, a, false, b0, (short)0, acc[0], false, false);
            acc[1] = __builtin_amdgcn_wmma_f32_16x16x32_bf16(
                false, a, false, b1, (short)0, acc[1], false, false);
        }
#pragma unroll
        for (int i = 0; i < 2; ++i) {
            const int col = n0 + (ntb + i) * 16 + ln;
#pragma unroll
            for (int r = 0; r < 8; ++r) {
                const int row = mt * 16 + r + 8 * g;   // batch index
                const float v = acc[i][r] + bf2f(XWt[(size_t)row * HDIM + col]);
                Hout[(size_t)row * HDIM + col] = f2bf(tanhf(v));
            }
        }
        // device-scope barrier across the 16 workgroups (monotonic counter)
        __threadfence();
        __syncthreads();
        if (tid == 0)
            __hip_atomic_fetch_add(cnt, 1u, __ATOMIC_RELEASE,
                                   __HIP_MEMORY_SCOPE_AGENT);
        const unsigned target = 16u * (unsigned)(t + 1);
        while (__hip_atomic_load(cnt, __ATOMIC_ACQUIRE,
                                 __HIP_MEMORY_SCOPE_AGENT) < target)
            __builtin_amdgcn_s_sleep(1);
        __syncthreads();
    }
}

// ---------- Phase 3: out = h_last @ W_out^T + b_out  (fp32 out) ----------------
__global__ void __launch_bounds__(256)
out_gemm(const unsigned short* __restrict__ Hlast,
         const unsigned short* __restrict__ Wout,
         const float* __restrict__ bout, float* __restrict__ out) {
    const int lane = threadIdx.x & 31;
    const int wave = threadIdx.x >> 5;
    const int g = lane >> 4, ln = lane & 15;
    const int mt = wave & 3, ntb = (wave >> 2) * 2;
    const int n0 = blockIdx.x * 64;

    const unsigned short* arow  = Hlast + (size_t)(mt * 16 + ln) * HDIM;
    const unsigned short* brow0 = Wout + (size_t)(n0 + ntb * 16 + ln) * HDIM + 16 * g;
    const unsigned short* brow1 = brow0 + (size_t)16 * HDIM;

    v8f acc[2] = {};
#pragma unroll 8
    for (int k0 = 0; k0 < HDIM; k0 += 32) {
        v16bf a  = load_afrag_bf16(arow, k0, g);
        v16bf b0 = load_bfrag(brow0 + k0);
        v16bf b1 = load_bfrag(brow1 + k0);
        acc[0] = __builtin_amdgcn_wmma_f32_16x16x32_bf16(
            false, a, false, b0, (short)0, acc[0], false, false);
        acc[1] = __builtin_amdgcn_wmma_f32_16x16x32_bf16(
            false, a, false, b1, (short)0, acc[1], false, false);
    }
#pragma unroll
    for (int i = 0; i < 2; ++i) {
        const int col = n0 + (ntb + i) * 16 + ln;
        const float bias = bout[col];
#pragma unroll
        for (int r = 0; r < 8; ++r) {
            const int row = mt * 16 + r + 8 * g;
            out[(size_t)row * ODIM + col] = acc[i][r] + bias;
        }
    }
}

// ---------- host-side orchestration -------------------------------------------
extern "C" void kernel_launch(void* const* d_in, const int* in_sizes, int n_in,
                              void* d_out, int out_size, void* d_ws, size_t ws_size,
                              hipStream_t stream) {
    (void)in_sizes; (void)n_in; (void)out_size; (void)ws_size;

    const float* X    = (const float*)d_in[0];   // [B,T,IDIM]
    const float* Wih  = (const float*)d_in[1];   // [HDIM,IDIM]
    const float* Whh  = (const float*)d_in[2];   // [HDIM,HDIM]
    const float* bh   = (const float*)d_in[3];   // [HDIM]
    const float* Wout = (const float*)d_in[4];   // [ODIM,HDIM]
    const float* bout = (const float*)d_in[5];   // [ODIM]
    float* out = (float*)d_out;                  // [B,ODIM] fp32

    // Workspace layout (units: ushort/bf16)
    unsigned short* ws = (unsigned short*)d_ws;
    const size_t nXW = (size_t)TT * BZ * HDIM;               // 256 MiB
    unsigned short* XW      = ws;
    unsigned short* Whh_bf  = XW + nXW;
    unsigned short* Wih_bf  = Whh_bf + (size_t)HDIM * HDIM;
    unsigned short* Wout_bf = Wih_bf + (size_t)HDIM * IDIM;
    unsigned short* Hbuf    = Wout_bf + (size_t)ODIM * HDIM; // 2 x [64x1024]
    unsigned int*   cnt     = (unsigned int*)(Hbuf + 2 * (size_t)BZ * HDIM);

    // 1) weights fp32 -> bf16 (stay L2-resident); zero h0 + barrier counter
    cvt_f32_to_bf16<<<(HDIM * IDIM + 255) / 256, 256, 0, stream>>>(Wih, Wih_bf, HDIM * IDIM);
    cvt_f32_to_bf16<<<(HDIM * HDIM + 255) / 256, 256, 0, stream>>>(Whh, Whh_bf, HDIM * HDIM);
    cvt_f32_to_bf16<<<(ODIM * HDIM + 255) / 256, 256, 0, stream>>>(Wout, Wout_bf, ODIM * HDIM);
    zero_u16<<<(BZ * HDIM + 255) / 256, 256, 0, stream>>>(Hbuf, BZ * HDIM);
    zero_u32<<<1, 256, 0, stream>>>(cnt, 1);

    // 2) parallel bulk GEMM (275 GFLOP on the WMMA path, async-LDS staged B)
    xw_gemm<<<dim3(16, 1024), 256, 0, stream>>>(X, Wih_bf, bh, XW);

    // 3) the whole 2048-step scan in ONE persistent kernel (16 co-resident WGs)
    rnn_scan<<<16, 256, 0, stream>>>(Whh_bf, XW, Hbuf, cnt);

    // 4) output projection from h_last (slot 0 after even T)
    out_gemm<<<8, 256, 0, stream>>>(Hbuf, Wout_bf, bout, out);
}
